// BertTokenEmbeddingModel_85899346302
// MI455X (gfx1250) — matrix operand order repaired and checked by reading, not compiled, not run on hardware
//
#include <hip/hip_runtime.h>
#include <hip/hip_bf16.h>
#include <math.h>

// ---------------------------------------------------------------------------
// BERT forward (B=16,S=512,H=768,NH=12,DH=64,L=2) + segment-mean pooling.
// All GEMMs (QKV, attention QK^T, PV, Wo, FF1, FF2) run on CDNA5 WMMA:
//   v_wmma_f32_16x16x32_bf16  (bf16 A/B, f32 accumulate)
// Global->LDS tile staging uses GLOBAL_LOAD_ASYNC_TO_LDS_B128 when the
// toolchain exposes the builtin (ASYNCcnt-tracked), else vector copies.
// ---------------------------------------------------------------------------

#define Bdim 16
#define Sdim 512
#define Hdim 768
#define NHdim 12
#define DHdim 64

typedef unsigned short u16;
typedef unsigned int u32;
typedef __attribute__((ext_vector_type(16))) __bf16 v16bf;
typedef __attribute__((ext_vector_type(8)))  float  v8f;
typedef __attribute__((ext_vector_type(4)))  int    v4i;

union Frag { v16bf v; uint4 q[2]; };

__device__ __forceinline__ u16 f2bf(float f) {
  unsigned int u = __float_as_uint(f);
  u += 0x7fffu + ((u >> 16) & 1u);   // round to nearest even
  return (u16)(u >> 16);
}

__device__ __forceinline__ float geluf(float v) {
  return 0.5f * v * (1.0f + erff(v * 0.70710678118654752f));
}

// ---------------------------------------------------------------------------
// async global->LDS 16-byte copy (per-lane), with synchronous fallback
// ---------------------------------------------------------------------------
#if defined(__HIP_DEVICE_COMPILE__) && \
    __has_builtin(__builtin_amdgcn_global_load_async_to_lds_b128)
#define ATHENA_ASYNC_LDS 1
#endif

__device__ __forceinline__ void cp_async16(u16* ldst, const u16* gsrc) {
#ifdef ATHENA_ASYNC_LDS
  __builtin_amdgcn_global_load_async_to_lds_b128(
      (__attribute__((address_space(1))) v4i*)gsrc,
      (__attribute__((address_space(3))) v4i*)ldst, 0, 0);
#else
  *reinterpret_cast<uint4*>(ldst) = *reinterpret_cast<const uint4*>(gsrc);
#endif
}

__device__ __forceinline__ void cp_async_wait() {
#ifdef ATHENA_ASYNC_LDS
#if __has_builtin(__builtin_amdgcn_s_wait_asynccnt)
  __builtin_amdgcn_s_wait_asynccnt(0);
#else
  asm volatile("s_wait_asynccnt 0" ::: "memory");
#endif
#endif
}

// ---------------------------------------------------------------------------
// fp32 -> bf16 bulk convert (for weights)
// ---------------------------------------------------------------------------
__global__ __launch_bounds__(256) void f32bf_kernel(const float* __restrict__ src,
                                                    u16* __restrict__ dst, int n) {
  int i = blockIdx.x * 256 + threadIdx.x;
  if (i < n) dst[i] = f2bf(src[i]);
}

// ---------------------------------------------------------------------------
// block-wide sum over 256 threads
// ---------------------------------------------------------------------------
__device__ __forceinline__ float block_sum256(float v, float* red) {
  int t = threadIdx.x;
  red[t] = v;
  __syncthreads();
  for (int off = 128; off > 0; off >>= 1) {
    if (t < off) red[t] += red[t + off];
    __syncthreads();
  }
  float r = red[0];
  __syncthreads();
  return r;
}

// ---------------------------------------------------------------------------
// x = LN(emb[id] + pos); writes fp32 x and bf16 mirror. one block per token.
// ---------------------------------------------------------------------------
__global__ __launch_bounds__(256)
void embed_ln_kernel(const int* __restrict__ tok, const float* __restrict__ emb,
                     const float* __restrict__ pos, const float* __restrict__ g,
                     const float* __restrict__ be, float* __restrict__ x,
                     u16* __restrict__ xb) {
  __shared__ float red[256];
  int row = blockIdx.x;            // b*S + s
  int s = row & (Sdim - 1);
  int id = tok[row * 2 + 1];
  float v[3];
  int hh[3];
#pragma unroll
  for (int i = 0; i < 3; ++i) {
    hh[i] = threadIdx.x + i * 256;
    v[i] = emb[(size_t)id * Hdim + hh[i]] + pos[(size_t)s * Hdim + hh[i]];
  }
  float mean = block_sum256(v[0] + v[1] + v[2], red) * (1.0f / Hdim);
  float d0 = v[0] - mean, d1 = v[1] - mean, d2 = v[2] - mean;
  float var = block_sum256(d0 * d0 + d1 * d1 + d2 * d2, red) * (1.0f / Hdim);
  float rstd = rsqrtf(var + 1e-12f);
#pragma unroll
  for (int i = 0; i < 3; ++i) {
    float o = (v[i] - mean) * rstd * g[hh[i]] + be[hh[i]];
    x[(size_t)row * Hdim + hh[i]] = o;
    xb[(size_t)row * Hdim + hh[i]] = f2bf(o);
  }
}

// ---------------------------------------------------------------------------
// x = LN(x + y); in-place fp32 update + bf16 mirror. one block per token.
// ---------------------------------------------------------------------------
__global__ __launch_bounds__(256)
void residual_ln_kernel(float* __restrict__ x, const float* __restrict__ y,
                        const float* __restrict__ g, const float* __restrict__ be,
                        u16* __restrict__ xb) {
  __shared__ float red[256];
  size_t row = blockIdx.x;
  float v[3];
  int hh[3];
#pragma unroll
  for (int i = 0; i < 3; ++i) {
    hh[i] = threadIdx.x + i * 256;
    v[i] = x[row * Hdim + hh[i]] + y[row * Hdim + hh[i]];
  }
  float mean = block_sum256(v[0] + v[1] + v[2], red) * (1.0f / Hdim);
  float d0 = v[0] - mean, d1 = v[1] - mean, d2 = v[2] - mean;
  float var = block_sum256(d0 * d0 + d1 * d1 + d2 * d2, red) * (1.0f / Hdim);
  float rstd = rsqrtf(var + 1e-12f);
#pragma unroll
  for (int i = 0; i < 3; ++i) {
    float o = (v[i] - mean) * rstd * g[hh[i]] + be[hh[i]];
    x[row * Hdim + hh[i]] = o;
    xb[row * Hdim + hh[i]] = f2bf(o);
  }
}

// ---------------------------------------------------------------------------
// WMMA GEMM: C[M,N] = A_bf16[M,K] @ B_bf16[K,N] + bias   (row-major)
// 128x128 tile / block (256 threads = 8 waves; each wave 32x64 via 2x4 accs).
// LDS: A tile [128][32] (M-major), B tile transposed [128 n][32 k]; both give
// fragment loads as 2x ds_read_b128 matching the ISA wave32 VGPR layouts.
// ACT: 0 = none, 1 = exact GELU.  OUTB: 1 = store bf16, 0 = store fp32.
// ---------------------------------------------------------------------------
#define LSTR 40  // padded LDS row stride (u16 units); 80B keeps 16B alignment

template <int ACT, int OUTB>
__global__ __launch_bounds__(256)
void gemm_wmma(const u16* __restrict__ A, const u16* __restrict__ Bw,
               const float* __restrict__ bias, float* __restrict__ Cf,
               u16* __restrict__ Cb, int M, int N, int K) {
  __shared__ u16 As[128 * LSTR];
  __shared__ u16 Bs[128 * LSTR];
  const int t = threadIdx.x;
  const int wave = t >> 5, lane = t & 31;
  const int wm = wave & 3, wn = wave >> 2;        // wave tile: M 32, N 64
  const int lh = lane >> 4, ll = lane & 15;
  const int m0 = blockIdx.y * 128, n0 = blockIdx.x * 128;

  v8f acc[2][4];
#pragma unroll
  for (int mi = 0; mi < 2; ++mi)
#pragma unroll
    for (int ni = 0; ni < 4; ++ni)
#pragma unroll
      for (int i = 0; i < 8; ++i) acc[mi][ni][i] = 0.0f;

  for (int k0 = 0; k0 < K; k0 += 32) {
    __syncthreads();
    // ---- stage A tile: 128 rows x 32 k (bf16); async b128 per lane --------
#pragma unroll
    for (int i = 0; i < 2; ++i) {
      int gidx = t + i * 256;               // 0..511 groups of 8 elems
      int r = gidx >> 2, c = (gidx & 3) * 8;
      const u16* src = A + (size_t)(m0 + r) * K + k0 + c;
      cp_async16(&As[r * LSTR + c], src);
      if (k0 + 32 < K) __builtin_prefetch(src + 32, 0, 1);  // global_prefetch_b8
    }
    // ---- stage B tile transposed: Bs[n][k]; pack k-pairs -> ds_store_b32 --
    {
      int kk = (t >> 4) * 2;                // even k row 0..30
      int nn = (t & 15) * 8;                // n group of 8
      const u16* src0 = Bw + (size_t)(k0 + kk) * N + n0 + nn;
      const u16* src1 = src0 + N;
      uint4 r0 = *reinterpret_cast<const uint4*>(src0);
      uint4 r1 = *reinterpret_cast<const uint4*>(src1);
      const u16* e0 = reinterpret_cast<const u16*>(&r0);
      const u16* e1 = reinterpret_cast<const u16*>(&r1);
#pragma unroll
      for (int j = 0; j < 8; ++j) {
        u32 d = (u32)e0[j] | ((u32)e1[j] << 16);
        *reinterpret_cast<u32*>(&Bs[(nn + j) * LSTR + kk]) = d;
      }
      if (k0 + 32 < K) __builtin_prefetch(src0 + (size_t)32 * N, 0, 1);
    }
    cp_async_wait();
    __syncthreads();

    // ---- fragments (ISA layouts) ------------------------------------------
    Frag afr[2], bfr[4];
#pragma unroll
    for (int mi = 0; mi < 2; ++mi) {
      int row = wm * 32 + mi * 16 + ll;     // lanes 0-15 & 16-31: same rows
      afr[mi].q[0] = *reinterpret_cast<const uint4*>(&As[row * LSTR + lh * 8]);
      afr[mi].q[1] = *reinterpret_cast<const uint4*>(&As[row * LSTR + 16 + lh * 8]);
    }
#pragma unroll
    for (int ni = 0; ni < 4; ++ni) {
      int col = wn * 64 + ni * 16 + ll;     // lane -> column
      bfr[ni].q[0] = *reinterpret_cast<const uint4*>(&Bs[col * LSTR + lh * 8]);
      bfr[ni].q[1] = *reinterpret_cast<const uint4*>(&Bs[col * LSTR + 16 + lh * 8]);
    }
#pragma unroll
    for (int mi = 0; mi < 2; ++mi)
#pragma unroll
      for (int ni = 0; ni < 4; ++ni)
        acc[mi][ni] = __builtin_amdgcn_wmma_f32_16x16x32_bf16(
            false, afr[mi].v, false, bfr[ni].v, (short)0, acc[mi][ni], false, false);
  }

  // ---- epilogue: bias (+GELU), store fp32 or bf16 --------------------------
#pragma unroll
  for (int mi = 0; mi < 2; ++mi)
#pragma unroll
    for (int ni = 0; ni < 4; ++ni) {
      int col = n0 + wn * 64 + ni * 16 + ll;
      float bv = bias[col];
#pragma unroll
      for (int i = 0; i < 8; ++i) {
        int row = m0 + wm * 32 + mi * 16 + lh * 8 + i;
        float v = acc[mi][ni][i] + bv;
        if (ACT == 1) v = geluf(v);
        if (OUTB)
          Cb[(size_t)row * N + col] = f2bf(v);
        else
          Cf[(size_t)row * N + col] = v;
      }
    }
}

// ---------------------------------------------------------------------------
// Fused flash attention, one block per (q-tile 64, head, batch).
// 128 threads = 4 waves; wave w owns q rows [w*16, w*16+16).
// scores = Q K^T / 8 -> online softmax -> O += P V, all via bf16 WMMA.
// ---------------------------------------------------------------------------
#define ASTR 72  // LDS row stride (u16) for 64-wide tiles, keeps 16B alignment

__global__ __launch_bounds__(128)
void attn_wmma(const u16* __restrict__ qkv, u16* __restrict__ ctx) {
  __shared__ u16 Qs[64 * ASTR];
  __shared__ u16 Ks[64 * ASTR];  // [kv][dh]
  __shared__ u16 Vs[64 * ASTR];  // [dh][kv]  (transposed on load)
  __shared__ u16 Ps[64 * ASTR];  // [q][kv]   bf16 probabilities

  const int b = blockIdx.z, h = blockIdx.y, q0 = blockIdx.x * 64;
  const int t = threadIdx.x, wave = t >> 5, lane = t & 31;
  const int lh = lane >> 4, ll = lane & 15;
  const size_t rs = 3 * Hdim;                       // qkv row stride (2304)
  const size_t base = (size_t)b * Sdim * rs;

  // ---- load Q tile (64 x 64 bf16), async ----------------------------------
#pragma unroll
  for (int i = 0; i < 4; ++i) {
    int gidx = t + i * 128;                          // 512 groups of 8
    int r = gidx >> 3, c = (gidx & 7) * 8;
    cp_async16(&Qs[r * ASTR + c],
               qkv + base + (size_t)(q0 + r) * rs + h * DHdim + c);
  }
  cp_async_wait();
  __syncthreads();

  // Q fragments for this wave (2 k-steps over DH=64)
  Frag aq[2];
  {
    int row = wave * 16 + ll;
    aq[0].q[0] = *reinterpret_cast<const uint4*>(&Qs[row * ASTR + lh * 8]);
    aq[0].q[1] = *reinterpret_cast<const uint4*>(&Qs[row * ASTR + 16 + lh * 8]);
    aq[1].q[0] = *reinterpret_cast<const uint4*>(&Qs[row * ASTR + 32 + lh * 8]);
    aq[1].q[1] = *reinterpret_cast<const uint4*>(&Qs[row * ASTR + 48 + lh * 8]);
  }

  v8f o[4];
  float mrun[8], lrun[8];
#pragma unroll
  for (int ni = 0; ni < 4; ++ni)
#pragma unroll
    for (int i = 0; i < 8; ++i) o[ni][i] = 0.0f;
#pragma unroll
  for (int i = 0; i < 8; ++i) { mrun[i] = -3.4e38f; lrun[i] = 0.0f; }

  for (int j0 = 0; j0 < Sdim; j0 += 64) {
    __syncthreads();
    // ---- stage K chunk [kv][dh], async ------------------------------------
#pragma unroll
    for (int i = 0; i < 4; ++i) {
      int gidx = t + i * 128;
      int r = gidx >> 3, c = (gidx & 7) * 8;
      cp_async16(&Ks[r * ASTR + c],
                 qkv + base + (size_t)(j0 + r) * rs + Hdim + h * DHdim + c);
    }
    // ---- stage V chunk transposed: Vs[dh][kv]; pack kv-pairs -> b32 -------
#pragma unroll
    for (int i = 0; i < 4; ++i) {
      int gidx = t + i * 128;                        // 512 groups
      int kv2 = (gidx >> 4) * 2;                     // even kv row 0..62
      int c4 = (gidx & 15) * 4;                      // dh group of 4
      const u16* s0 = qkv + base + (size_t)(j0 + kv2) * rs + 2 * Hdim + h * DHdim + c4;
      const u16* s1 = s0 + rs;
      uint2 a0 = *reinterpret_cast<const uint2*>(s0);
      uint2 a1 = *reinterpret_cast<const uint2*>(s1);
      const u16* e0 = reinterpret_cast<const u16*>(&a0);
      const u16* e1 = reinterpret_cast<const u16*>(&a1);
#pragma unroll
      for (int j = 0; j < 4; ++j) {
        u32 d = (u32)e0[j] | ((u32)e1[j] << 16);
        *reinterpret_cast<u32*>(&Vs[(c4 + j) * ASTR + kv2]) = d;
      }
    }
    cp_async_wait();
    __syncthreads();

    // ---- scores: 16 q rows x 64 kv cols -----------------------------------
    v8f sc[4];
#pragma unroll
    for (int ni = 0; ni < 4; ++ni)
#pragma unroll
      for (int i = 0; i < 8; ++i) sc[ni][i] = 0.0f;
#pragma unroll
    for (int ks = 0; ks < 2; ++ks) {
      Frag bk[4];
#pragma unroll
      for (int ni = 0; ni < 4; ++ni) {
        int col = ni * 16 + ll;
        bk[ni].q[0] = *reinterpret_cast<const uint4*>(&Ks[col * ASTR + ks * 32 + lh * 8]);
        bk[ni].q[1] = *reinterpret_cast<const uint4*>(&Ks[col * ASTR + ks * 32 + 16 + lh * 8]);
      }
#pragma unroll
      for (int ni = 0; ni < 4; ++ni)
        sc[ni] = __builtin_amdgcn_wmma_f32_16x16x32_bf16(
            false, aq[ks].v, false, bk[ni].v, (short)0, sc[ni], false, false);
    }

    // ---- online softmax (scale = 1/sqrt(64) = 0.125) ----------------------
#pragma unroll
    for (int i = 0; i < 8; ++i) {
      float mx = -3.4e38f;
#pragma unroll
      for (int ni = 0; ni < 4; ++ni) mx = fmaxf(mx, sc[ni][i] * 0.125f);
      mx = fmaxf(mx, __shfl_xor(mx, 1, 16));
      mx = fmaxf(mx, __shfl_xor(mx, 2, 16));
      mx = fmaxf(mx, __shfl_xor(mx, 4, 16));
      mx = fmaxf(mx, __shfl_xor(mx, 8, 16));
      float mn = fmaxf(mrun[i], mx);
      float corr = __expf(mrun[i] - mn);
      float rsum = 0.0f;
      int prow = wave * 16 + lh * 8 + i;
#pragma unroll
      for (int ni = 0; ni < 4; ++ni) {
        float p = __expf(sc[ni][i] * 0.125f - mn);
        rsum += p;
        Ps[prow * ASTR + ni * 16 + ll] = f2bf(p);
      }
      rsum += __shfl_xor(rsum, 1, 16);
      rsum += __shfl_xor(rsum, 2, 16);
      rsum += __shfl_xor(rsum, 4, 16);
      rsum += __shfl_xor(rsum, 8, 16);
      lrun[i] = lrun[i] * corr + rsum;
#pragma unroll
      for (int ni = 0; ni < 4; ++ni) o[ni][i] *= corr;
      mrun[i] = mn;
    }
    // Ps rows are wave-private; per-wave LDS ordering suffices (no barrier).

    // ---- O += P V -----------------------------------------------------------
#pragma unroll
    for (int ks = 0; ks < 2; ++ks) {
      Frag ap;
      int row = wave * 16 + ll;
      ap.q[0] = *reinterpret_cast<const uint4*>(&Ps[row * ASTR + ks * 32 + lh * 8]);
      ap.q[1] = *reinterpret_cast<const uint4*>(&Ps[row * ASTR + ks * 32 + 16 + lh * 8]);
      Frag bv[4];
#pragma unroll
      for (int ni = 0; ni < 4; ++ni) {
        int col = ni * 16 + ll;                      // dh column
        bv[ni].q[0] = *reinterpret_cast<const uint4*>(&Vs[col * ASTR + ks * 32 + lh * 8]);
        bv[ni].q[1] = *reinterpret_cast<const uint4*>(&Vs[col * ASTR + ks * 32 + 16 + lh * 8]);
      }
#pragma unroll
      for (int ni = 0; ni < 4; ++ni)
        o[ni] = __builtin_amdgcn_wmma_f32_16x16x32_bf16(
            false, ap.v, false, bv[ni].v, (short)0, o[ni], false, false);
    }
  }

  // ---- normalize + store ctx (bf16) ---------------------------------------
#pragma unroll
  for (int ni = 0; ni < 4; ++ni)
#pragma unroll
    for (int i = 0; i < 8; ++i) {
      int row = q0 + wave * 16 + lh * 8 + i;
      float inv = lrun[i] > 0.0f ? 1.0f / lrun[i] : 0.0f;
      ctx[(size_t)(b * Sdim + row) * Hdim + h * DHdim + ni * 16 + ll] = f2bf(o[ni][i] * inv);
    }
}

// ---------------------------------------------------------------------------
// Segment pooling. Groups are contiguous in s; masked tokens are a prefix of
// each group (padding is a suffix of the sequence).
// ---------------------------------------------------------------------------
__global__ void seg_scan_kernel(const int* __restrict__ tok, int* __restrict__ gstart,
                                int* __restrict__ gcnt, int* __restrict__ ngr) {
  int b = blockIdx.x;
  if (threadIdx.x != 0) return;
  int prev = -2147483647, g = -1;
  for (int s = 0; s < Sdim; ++s) {
    int wid = tok[(b * Sdim + s) * 2 + 0];
    int id = tok[(b * Sdim + s) * 2 + 1];
    if (s == 0 || wid != prev) {
      ++g;
      gstart[b * Sdim + g] = s;
      gcnt[b * Sdim + g] = 0;
    }
    prev = wid;
    if (id != 0) gcnt[b * Sdim + g] += 1;   // PAD == 0
  }
  ngr[b] = g + 1;
}

__global__ __launch_bounds__(256)
void seg_mean_kernel(const float* __restrict__ x, const int* __restrict__ gstart,
                     const int* __restrict__ gcnt, const int* __restrict__ ngr,
                     float* __restrict__ out) {
  int row = blockIdx.x;                 // b*S + g
  int b = row >> 9, g = row & (Sdim - 1);
  bool valid = (g < ngr[b]) && (gcnt[row] > 0);
  int st = valid ? gstart[row] : 0;
  int cn = valid ? gcnt[row] : 0;
  float inv = valid ? 1.0f / (float)cn : 0.0f;
  for (int hh = threadIdx.x; hh < Hdim; hh += 256) {
    float acc = 0.0f;
    for (int s2 = 0; s2 < cn; ++s2)
      acc += x[(size_t)(b * Sdim + st + s2) * Hdim + hh];
    out[(size_t)row * Hdim + hh] = acc * inv;   // 0 where group invalid/empty
  }
}

// ---------------------------------------------------------------------------
// launcher — workspace needs ~170 MB
// ---------------------------------------------------------------------------
extern "C" void kernel_launch(void* const* d_in, const int* in_sizes, int n_in,
                              void* d_out, int out_size, void* d_ws, size_t ws_size,
                              hipStream_t stream) {
  (void)in_sizes; (void)n_in; (void)out_size; (void)ws_size;
  const int* tok = (const int*)d_in[0];
  const float* emb = (const float*)d_in[1];
  const float* pos = (const float*)d_in[2];
  const float* ln_emb_g = (const float*)d_in[3];
  const float* ln_emb_b = (const float*)d_in[4];
  const float* Wqkv = (const float*)d_in[5];
  const float* bqkv = (const float*)d_in[6];
  const float* Wo = (const float*)d_in[7];
  const float* bo = (const float*)d_in[8];
  const float* ln1_g = (const float*)d_in[9];
  const float* ln1_b = (const float*)d_in[10];
  const float* Wff1 = (const float*)d_in[11];
  const float* bff1 = (const float*)d_in[12];
  const float* Wff2 = (const float*)d_in[13];
  const float* bff2 = (const float*)d_in[14];
  const float* ln2_g = (const float*)d_in[15];
  const float* ln2_b = (const float*)d_in[16];
  float* out = (float*)d_out;

  const size_t R = (size_t)Bdim * Sdim;   // 8192 token rows
  char* ws = (char*)d_ws;
  size_t off = 0;
  auto take = [&](size_t bytes) -> char* {
    char* p = ws + off;
    off += (bytes + 255) & ~(size_t)255;
    return p;
  };
  float* xf = (float*)take(R * Hdim * 4);          // fp32 residual stream
  u16* xb = (u16*)take(R * Hdim * 2);              // bf16 mirror of x
  u16* qkvb = (u16*)take(R * 3 * Hdim * 2);        // bf16 qkv
  u16* ctxb = (u16*)take(R * Hdim * 2);            // bf16 attention context
  float* yf = (float*)take(R * Hdim * 4);          // fp32 gemm output
  u16* ff1b = (u16*)take(R * 4 * Hdim * 2);        // bf16 FF hidden
  u16* wb = (u16*)take((size_t)Hdim * 4 * Hdim * 2);  // bf16 weight staging
  int* gst = (int*)take(R * 4);
  int* gcn = (int*)take(R * 4);
  int* ngr = (int*)take(Bdim * 4);

  embed_ln_kernel<<<(int)R, 256, 0, stream>>>(tok, emb, pos, ln_emb_g, ln_emb_b, xf, xb);

  for (int l = 0; l < 2; ++l) {
    // QKV projection (bf16 out)
    {
      int n = Hdim * 3 * Hdim;
      f32bf_kernel<<<(n + 255) / 256, 256, 0, stream>>>(Wqkv + (size_t)l * n, wb, n);
      gemm_wmma<0, 1><<<dim3(3 * Hdim / 128, (int)R / 128), 256, 0, stream>>>(
          xb, wb, bqkv + l * 3 * Hdim, nullptr, qkvb, (int)R, 3 * Hdim, Hdim);
    }
    // fused attention -> ctx (bf16)
    attn_wmma<<<dim3(Sdim / 64, NHdim, Bdim), 128, 0, stream>>>(qkvb, ctxb);
    // output projection (fp32 out) + residual LN
    {
      int n = Hdim * Hdim;
      f32bf_kernel<<<(n + 255) / 256, 256, 0, stream>>>(Wo + (size_t)l * n, wb, n);
      gemm_wmma<0, 0><<<dim3(Hdim / 128, (int)R / 128), 256, 0, stream>>>(
          ctxb, wb, bo + l * Hdim, yf, nullptr, (int)R, Hdim, Hdim);
      residual_ln_kernel<<<(int)R, 256, 0, stream>>>(xf, yf, ln1_g + l * Hdim, ln1_b + l * Hdim, xb);
    }
    // FFN up (GELU, bf16 out)
    {
      int n = Hdim * 4 * Hdim;
      f32bf_kernel<<<(n + 255) / 256, 256, 0, stream>>>(Wff1 + (size_t)l * n, wb, n);
      gemm_wmma<1, 1><<<dim3(4 * Hdim / 128, (int)R / 128), 256, 0, stream>>>(
          xb, wb, bff1 + l * 4 * Hdim, nullptr, ff1b, (int)R, 4 * Hdim, Hdim);
    }
    // FFN down (fp32 out) + residual LN
    {
      int n = 4 * Hdim * Hdim;
      f32bf_kernel<<<(n + 255) / 256, 256, 0, stream>>>(Wff2 + (size_t)l * n, wb, n);
      gemm_wmma<0, 0><<<dim3(Hdim / 128, (int)R / 128), 256, 0, stream>>>(
          ff1b, wb, bff2 + l * Hdim, yf, nullptr, (int)R, Hdim, 4 * Hdim);
      residual_ln_kernel<<<(int)R, 256, 0, stream>>>(xf, yf, ln2_g + l * Hdim, ln2_b + l * Hdim, xb);
    }
  }

  seg_scan_kernel<<<Bdim, 32, 0, stream>>>(tok, gst, gcn, ngr);
  seg_mean_kernel<<<(int)R, 256, 0, stream>>>(xf, gst, gcn, ngr, out);
}
